// Attention_31473520345102
// MI455X (gfx1250) — compile-verified, matrix operand order checked by reading, not run on hardware
//
#include <hip/hip_runtime.h>

#define DIM   768
#define NH    12
#define HD    64
#define NSEQ  4096
#define BATCH 2
#define MROWS (BATCH * NSEQ)   // 8192

typedef __attribute__((ext_vector_type(16))) __bf16 v16bf;
typedef __attribute__((ext_vector_type(8)))  __bf16 v8bf;
typedef __attribute__((ext_vector_type(8)))  float  v8f;

// ---- f32 -> bf16 round-to-nearest-even (pure bit ops) ----
__device__ __forceinline__ __bf16 tobf(float f) {
    unsigned u = __builtin_bit_cast(unsigned, f);
    unsigned r = (u + 0x7FFFu + ((u >> 16) & 1u)) >> 16;
    unsigned short s = (unsigned short)r;
    return __builtin_bit_cast(__bf16, s);
}

// ---- WMMA wrapper: D = A(16x32 bf16) * B(32x16 bf16) + C(16x16 f32) ----
__device__ __forceinline__ v8f wmma_bf16(v16bf a, v16bf b, v8f c) {
    return __builtin_amdgcn_wmma_f32_16x16x32_bf16(
        false, a, false, b, (short)0, c, false, false);
}

// ---- Async global->LDS 128-bit copy (CDNA5, tracked by ASYNCcnt) ----
__device__ __forceinline__ void async_load_b128(void* lds_ptr, const void* gptr) {
    unsigned lds_off = (unsigned)(uintptr_t)lds_ptr;   // generic addr low 32 = LDS offset
    unsigned long long ga = (unsigned long long)(uintptr_t)gptr;
    asm volatile("global_load_async_to_lds_b128 %0, %1, off"
                 :: "v"(lds_off), "v"(ga)
                 : "memory");
}
__device__ __forceinline__ void wait_async0() {
    asm volatile("s_wait_asynccnt 0x0" ::: "memory");
}

// ---- Fragment loader for the 16-bit 16x32 A layout (and mirrored B layout).
// Lane L (L<16): K chunks [0,8)+[16,24); lane L>=16: K chunks [8,16)+[24,32).
__device__ __forceinline__ v16bf load_frag(const __bf16* base, int ld,
                                           int row0, int k0, int lm, int half) {
    const __bf16* p0 = base + (size_t)(row0 + lm) * ld + k0 + half * 8;
    v8bf lo = *(const v8bf*)p0;
    v8bf hi = *(const v8bf*)(p0 + 16);
    v16bf r;
#pragma unroll
    for (int i = 0; i < 8; ++i) { r[i] = lo[i]; r[i + 8] = hi[i]; }
    return r;
}

// ====================== kernel 1: f32 -> bf16 convert ======================
__global__ void cvt_f32_bf16(const float* __restrict__ in, __bf16* __restrict__ out, int n) {
    int i = blockIdx.x * blockDim.x + threadIdx.x;
    if (i < n) out[i] = tobf(in[i]);
}

// ====================== kernel 2: QKV projection GEMM ======================
// Each wave: 32x64 output tile (2 A-frags x 4 B-frags -> 8 wmma / k-step).
__global__ __launch_bounds__(128) void qkv_gemm(
    const __bf16* __restrict__ X,      // [8192, 768]
    const __bf16* __restrict__ W,      // [2304, 768]
    __bf16* __restrict__ Q,            // [B,H,N,D], pre-scaled
    __bf16* __restrict__ K,            // [B,H,N,D]
    __bf16* __restrict__ VT) {         // [B,H,D,N]
    const int lane = threadIdx.x & 31;
    const int wave = threadIdx.x >> 5;
    const int lm = lane & 15, half = lane >> 4;
    const int row0 = blockIdx.x * 128 + wave * 32;
    const int col0 = blockIdx.y * 64;
    const float scale = 0.125f;  // HD^-0.5

    v8f acc[2][4] = {};
    for (int k = 0; k < DIM; k += 32) {
        v16bf a0 = load_frag(X, DIM, row0,      k, lm, half);
        v16bf a1 = load_frag(X, DIM, row0 + 16, k, lm, half);
#pragma unroll
        for (int t = 0; t < 4; ++t) {
            v16bf b = load_frag(W, DIM, col0 + t * 16, k, lm, half);
            acc[0][t] = wmma_bf16(a0, b, acc[0][t]);
            acc[1][t] = wmma_bf16(a1, b, acc[1][t]);
        }
    }
#pragma unroll
    for (int t = 0; t < 4; ++t) {
        const int o = col0 + t * 16 + lm;     // output feature (col = lane%16)
        const int tt = o / DIM;               // 0=q, 1=k, 2=v (uniform per block)
        const int rem = o % DIM;
        const int h = rem / HD, d = rem % HD;
#pragma unroll
        for (int mt = 0; mt < 2; ++mt)
#pragma unroll
            for (int r = 0; r < 8; ++r) {
                const int m = row0 + mt * 16 + r + 8 * half;
                const int b = m >> 12, n = m & (NSEQ - 1);
                const float v = acc[mt][t][r];
                const size_t bh = (size_t)b * NH + h;
                if (tt == 0)      Q[(bh * NSEQ + n) * HD + d] = tobf(v * scale);
                else if (tt == 1) K[(bh * NSEQ + n) * HD + d] = tobf(v);
                else              VT[(bh * HD + d) * NSEQ + n] = tobf(v);
            }
    }
}

// ====================== kernel 3: flash attention ======================
// 8 waves / block; block covers 128 query rows. K and V^T tiles (64 keys)
// are DOUBLE-BUFFERED in LDS: async copies for block j+1 are issued before
// computing on block j, so the async engine runs under the WMMA pipeline.
__global__ __launch_bounds__(256) void attn_kernel(
    const __bf16* __restrict__ Q, const __bf16* __restrict__ K,
    const __bf16* __restrict__ VT, __bf16* __restrict__ AOUT /* [B,N,C] */) {
    __shared__ __align__(16) __bf16 kt[2][64][64];      // [buf][key][d]   16KB
    __shared__ __align__(16) __bf16 vt[2][64][64];      // [buf][d][key]   16KB
    __shared__ __align__(16) __bf16 lds_p[8][16 * 64];  // per-wave P      16KB

    const int tid  = threadIdx.x;
    const int lane = tid & 31;
    const int wave = tid >> 5;
    const int lm = lane & 15, half = lane >> 4;
    const int bh = blockIdx.y;              // b*NH + h
    const int b = bh / NH, h = bh % NH;
    const int row0 = blockIdx.x * 128 + wave * 16;  // query row within (b,h)

    const __bf16* qh = Q + (size_t)bh * NSEQ * HD;
    const __bf16* kh = K + (size_t)bh * NSEQ * HD;
    const __bf16* vh = VT + (size_t)bh * HD * NSEQ;

    // per-thread staging coordinates: 512 16B chunks per 64x64 tile
    const int c0 = tid, c1 = tid + 256;
    const int sr0 = c0 >> 3, sf0 = (c0 & 7) * 8;
    const int sr1 = c1 >> 3, sf1 = (c1 & 7) * 8;

    // Q fragments (A-matrix, K-dim = D = 64 -> two 16x32 frags)
    const v16bf aq0 = load_frag(qh, HD, row0, 0, lm, half);
    const v16bf aq1 = load_frag(qh, HD, row0, 32, lm, half);

    v8f o[4] = {};
    float mrow[8], lrow[8];
#pragma unroll
    for (int r = 0; r < 8; ++r) { mrow[r] = -1e30f; lrow[r] = 0.0f; }

    // ---- prologue: stage key-block 0 into buffer 0 ----
    async_load_b128(&kt[0][sr0][sf0], kh + (size_t)sr0 * HD + sf0);
    async_load_b128(&kt[0][sr1][sf1], kh + (size_t)sr1 * HD + sf1);
    async_load_b128(&vt[0][sr0][sf0], vh + (size_t)sr0 * NSEQ + sf0);
    async_load_b128(&vt[0][sr1][sf1], vh + (size_t)sr1 * NSEQ + sf1);
    wait_async0();
    __syncthreads();

    for (int j0 = 0; j0 < NSEQ; j0 += 64) {
        const int cur = (j0 >> 6) & 1;
        const int nxt = cur ^ 1;

        // ---- issue async staging for the NEXT key block (overlaps compute) ----
        if (j0 + 64 < NSEQ) {
            const int jn = j0 + 64;
            async_load_b128(&kt[nxt][sr0][sf0], kh + (size_t)(jn + sr0) * HD + sf0);
            async_load_b128(&kt[nxt][sr1][sf1], kh + (size_t)(jn + sr1) * HD + sf1);
            async_load_b128(&vt[nxt][sr0][sf0], vh + (size_t)sr0 * NSEQ + jn + sf0);
            async_load_b128(&vt[nxt][sr1][sf1], vh + (size_t)sr1 * NSEQ + jn + sf1);
            if (jn + 64 < NSEQ)
                __builtin_prefetch(kh + (size_t)(jn + 64) * HD, 0, 0);
        }

        // ---- S = Q * K^T from LDS (16 rows x 64 keys) ----
        v8f s[4];
#pragma unroll
        for (int jt = 0; jt < 4; ++jt) {
            v16bf bk0 = load_frag(&kt[cur][0][0], 64, jt * 16, 0, lm, half);
            v16bf bk1 = load_frag(&kt[cur][0][0], 64, jt * 16, 32, lm, half);
            v8f z = {};
            z = wmma_bf16(aq0, bk0, z);
            z = wmma_bf16(aq1, bk1, z);
            s[jt] = z;
        }

        // ---- online softmax per row (row = r + 8*half, cols = lanes of 16-group) ----
#pragma unroll
        for (int r = 0; r < 8; ++r) {
            float vmax = fmaxf(fmaxf(s[0][r], s[1][r]), fmaxf(s[2][r], s[3][r]));
#pragma unroll
            for (int mk = 8; mk >= 1; mk >>= 1) vmax = fmaxf(vmax, __shfl_xor(vmax, mk, 32));
            const float mnew = fmaxf(mrow[r], vmax);
            const float corr = __expf(mrow[r] - mnew);
            float psum = 0.0f;
#pragma unroll
            for (int jt = 0; jt < 4; ++jt) {
                float p = __expf(s[jt][r] - mnew);
                s[jt][r] = p;
                psum += p;
            }
#pragma unroll
            for (int mk = 8; mk >= 1; mk >>= 1) psum += __shfl_xor(psum, mk, 32);
            lrow[r] = lrow[r] * corr + psum;
            mrow[r] = mnew;
#pragma unroll
            for (int dt = 0; dt < 4; ++dt) o[dt][r] *= corr;
        }

        // ---- C-layout -> A-layout for P via wave-private LDS (in-order per wave) ----
#pragma unroll
        for (int jt = 0; jt < 4; ++jt)
#pragma unroll
            for (int r = 0; r < 8; ++r)
                lds_p[wave][(r + 8 * half) * 64 + jt * 16 + lm] = tobf(s[jt][r]);
        const v16bf ap0 = load_frag(&lds_p[wave][0], 64, 0, 0, lm, half);
        const v16bf ap1 = load_frag(&lds_p[wave][0], 64, 0, 32, lm, half);

        // ---- O += P * V from LDS V^T tile ----
#pragma unroll
        for (int dt = 0; dt < 4; ++dt) {
            v16bf bv0 = load_frag(&vt[cur][0][0], 64, dt * 16, 0, lm, half);
            v16bf bv1 = load_frag(&vt[cur][0][0], 64, dt * 16, 32, lm, half);
            o[dt] = wmma_bf16(ap0, bv0, o[dt]);
            o[dt] = wmma_bf16(ap1, bv1, o[dt]);
        }

        // ---- pipeline point: next buffer filled, everyone done with current ----
        wait_async0();
        __syncthreads();
    }

    // ---- normalize and store to [B,N,C] bf16 for the projection GEMM ----
#pragma unroll
    for (int r = 0; r < 8; ++r) {
        const float inv = 1.0f / lrow[r];
        const int n = row0 + r + 8 * half;
#pragma unroll
        for (int dt = 0; dt < 4; ++dt) {
            const int d = dt * 16 + lm;
            AOUT[((size_t)(b * NSEQ + n)) * DIM + h * HD + d] = tobf(o[dt][r] * inv);
        }
    }
}

// ====================== kernel 4: output projection GEMM ======================
__global__ __launch_bounds__(128) void proj_gemm(
    const __bf16* __restrict__ A,      // [8192, 768]
    const __bf16* __restrict__ W,      // [768, 768]
    const float* __restrict__ bias,    // [768]
    float* __restrict__ OUT) {         // [8192, 768] f32
    const int lane = threadIdx.x & 31;
    const int wave = threadIdx.x >> 5;
    const int lm = lane & 15, half = lane >> 4;
    const int row0 = blockIdx.x * 128 + wave * 32;
    const int col0 = blockIdx.y * 64;

    v8f acc[2][4] = {};
    for (int k = 0; k < DIM; k += 32) {
        v16bf a0 = load_frag(A, DIM, row0,      k, lm, half);
        v16bf a1 = load_frag(A, DIM, row0 + 16, k, lm, half);
#pragma unroll
        for (int t = 0; t < 4; ++t) {
            v16bf b = load_frag(W, DIM, col0 + t * 16, k, lm, half);
            acc[0][t] = wmma_bf16(a0, b, acc[0][t]);
            acc[1][t] = wmma_bf16(a1, b, acc[1][t]);
        }
    }
#pragma unroll
    for (int t = 0; t < 4; ++t) {
        const int oc = col0 + t * 16 + lm;
        const float bv = bias[oc];
#pragma unroll
        for (int mt = 0; mt < 2; ++mt)
#pragma unroll
            for (int r = 0; r < 8; ++r) {
                const int m = row0 + mt * 16 + r + 8 * half;
                OUT[(size_t)m * DIM + oc] = acc[mt][t][r] + bv;
            }
    }
}

// ====================== host-side launcher ======================
extern "C" void kernel_launch(void* const* d_in, const int* in_sizes, int n_in,
                              void* d_out, int out_size, void* d_ws, size_t ws_size,
                              hipStream_t stream) {
    const float* x      = (const float*)d_in[0];  // [2,4096,768]
    const float* w_qkv  = (const float*)d_in[1];  // [2304,768]
    const float* w_proj = (const float*)d_in[2];  // [768,768]
    const float* b_proj = (const float*)d_in[3];  // [768]
    float* out = (float*)d_out;

    char* ws = (char*)d_ws;
    size_t off = 0;
    auto alloc = [&](size_t bytes) -> void* {
        void* p = ws + off;
        off = (off + bytes + 255) & ~(size_t)255;
        return p;
    };
    __bf16* xb     = (__bf16*)alloc((size_t)MROWS * DIM * 2);
    __bf16* wqkvb  = (__bf16*)alloc((size_t)3 * DIM * DIM * 2);
    __bf16* wprojb = (__bf16*)alloc((size_t)DIM * DIM * 2);
    __bf16* qb     = (__bf16*)alloc((size_t)MROWS * DIM * 2);
    __bf16* kb     = (__bf16*)alloc((size_t)MROWS * DIM * 2);
    __bf16* vtb    = (__bf16*)alloc((size_t)MROWS * DIM * 2);
    __bf16* aoutb  = (__bf16*)alloc((size_t)MROWS * DIM * 2);

    const int n_x  = MROWS * DIM;
    const int n_wq = 3 * DIM * DIM;
    const int n_wp = DIM * DIM;
    cvt_f32_bf16<<<(n_x  + 255) / 256, 256, 0, stream>>>(x,      xb,     n_x);
    cvt_f32_bf16<<<(n_wq + 255) / 256, 256, 0, stream>>>(w_qkv,  wqkvb,  n_wq);
    cvt_f32_bf16<<<(n_wp + 255) / 256, 256, 0, stream>>>(w_proj, wprojb, n_wp);

    qkv_gemm<<<dim3(MROWS / 128, (3 * DIM) / 64), 128, 0, stream>>>(xb, wqkvb, qb, kb, vtb);
    attn_kernel<<<dim3(NSEQ / 128, BATCH * NH), 256, 0, stream>>>(qb, kb, vtb, aoutb);
    proj_gemm<<<dim3(MROWS / 128, DIM / 64), 128, 0, stream>>>(aoutb, wprojb, b_proj, out);
}
